// EdgeConv_11854109737376
// MI455X (gfx1250) — matrix operand order repaired and checked by reading, not compiled, not run on hardware
//
#include <hip/hip_runtime.h>
#include <cstdint>

#define DEVFN __device__ __forceinline__

typedef __attribute__((ext_vector_type(8)))  float  v8f;
typedef __attribute__((ext_vector_type(8)))  __bf16 v8bf;
typedef __attribute__((ext_vector_type(16))) __bf16 v16bf;

constexpr int   BATCH  = 8;
constexpr int   NPTS   = 4096;
constexpr int   NTOT   = BATCH * NPTS;   // 32768
constexpr int   CH     = 64;
constexpr int   KNN    = 20;
constexpr float BN_EPS = 1e-5f;
constexpr float FLT_BIG = 3.0e38f;

// ---------------------------------------------------------------- WMMA helpers
DEVFN v16bf bfcat(v8bf lo, v8bf hi) {
  return __builtin_shufflevector(lo, hi, 0,1,2,3,4,5,6,7,8,9,10,11,12,13,14,15);
}
// A fragment (16x32 bf16, row-major source, K contiguous):
// lane: m = lane&15, hi = lane>>4. VGPR halves at K = kb + hi*8 (+16).
DEVFN v16bf load_a(const __bf16* row, int kb, int hi) {
  v8bf lo = *reinterpret_cast<const v8bf*>(row + kb + hi * 8);
  v8bf hh = *reinterpret_cast<const v8bf*>(row + kb + 16 + hi * 8);
  return bfcat(lo, hh);
}
// B fragment (32x16 bf16): lane owns column N = lane&15; VGPR halves at K = kb + hi*8 (+16)
// contiguous when the "column" is a row-major vector of length 64 (x / W rows).
DEVFN v16bf load_b(const __bf16* col, int kb, int hi) {
  v8bf lo = *reinterpret_cast<const v8bf*>(col + kb + hi * 8);
  v8bf hh = *reinterpret_cast<const v8bf*>(col + kb + 16 + hi * 8);
  return bfcat(lo, hh);
}
DEVFN v8f wmma_bf16(v16bf a, v16bf b, v8f c) {
  return __builtin_amdgcn_wmma_f32_16x16x32_bf16(
      /*neg_a=*/false, a, /*neg_b=*/false, b,
      /*c_mod=*/(short)0, c, /*reuse_a=*/false, /*reuse_b=*/false);
}

// ---------------------------------------------------------------- async copy (CDNA5)
// One per-lane 16B async copy global->LDS; tracked on ASYNCcnt (§15.18, op 98).
DEVFN void async_copy_b128(const void* gsrc, void* ldst) {
  unsigned loff = (unsigned)(uintptr_t)ldst;  // LDS offset = addr[31:0]
  asm volatile("global_load_async_to_lds_b128 %0, %1, off"
               :: "v"(loff), "v"(gsrc) : "memory");
}
DEVFN void wait_async_le1() { asm volatile("s_wait_asynccnt 0x1" ::: "memory"); }
DEVFN void wait_async_le0() { asm volatile("s_wait_asynccnt 0x0" ::: "memory"); }

// ---------------------------------------------------------------- Kernel 1: point prep
// sq[n] = ||x_n||^2 (f32 exact), xb[n][c] = bf16(x[n][c])
__global__ void k_prep_points(const float* __restrict__ x,
                              __bf16* __restrict__ xb,
                              float* __restrict__ sq) {
  int n = blockIdx.x * blockDim.x + threadIdx.x;
  if (n >= NTOT) return;
  const float4* xr = reinterpret_cast<const float4*>(x + (size_t)n * CH);
  __bf16* xo = xb + (size_t)n * CH;
  float s = 0.f;
  #pragma unroll
  for (int i = 0; i < CH / 4; ++i) {
    float4 v = xr[i];
    s += v.x * v.x + v.y * v.y + v.z * v.z + v.w * v.w;
    xo[4*i+0] = (__bf16)v.x; xo[4*i+1] = (__bf16)v.y;
    xo[4*i+2] = (__bf16)v.z; xo[4*i+3] = (__bf16)v.w;
  }
  sq[n] = s;
}

// ---------------------------------------------------------------- Kernel 2: weight prep
// W is [64][128] row-major.  W1b[o][c] = W[o][c];  W2b[o][c] = W[o][64+c] - W[o][c]
// (h_e = x_i W1^T + x_j (W2-W1)^T; Linear bias cancels inside BatchNorm.)
__global__ void k_prep_weights(const float* __restrict__ W,
                               __bf16* __restrict__ W1b,
                               __bf16* __restrict__ W2b) {
  for (int idx = threadIdx.x; idx < CH * CH; idx += blockDim.x) {
    int o = idx >> 6, c = idx & 63;
    float w1 = W[o * 128 + c];
    float w2 = W[o * 128 + 64 + c];
    W1b[o * 64 + c] = (__bf16)w1;
    W2b[o * 64 + c] = (__bf16)(w2 - w1);
  }
}

// ---------------------------------------------------------------- Kernel 3: P/Q GEMM (WMMA)
// P = x @ W1^T, Q = x @ (W2-W1)^T.   One wave per 16-point tile; N=64, K=64.
__global__ void k_feat_gemm(const __bf16* __restrict__ xb,
                            const __bf16* __restrict__ W1b,
                            const __bf16* __restrict__ W2b,
                            float* __restrict__ P, float* __restrict__ Q) {
  int lane = threadIdx.x & 31;
  int m = lane & 15, hi = lane >> 4;
  int n0 = blockIdx.x * 16;
  const __bf16* arow = xb + (size_t)(n0 + m) * CH;
  v16bf a0 = load_a(arow, 0, hi);
  v16bf a1 = load_a(arow, 32, hi);
  #pragma unroll
  for (int nt = 0; nt < 4; ++nt) {
    int o = nt * 16 + (lane & 15);
    const __bf16* c1 = W1b + (size_t)o * CH;
    const __bf16* c2 = W2b + (size_t)o * CH;
    v8f accP = {}; v8f accQ = {};
    accP = wmma_bf16(a0, load_b(c1, 0, hi), accP);
    accP = wmma_bf16(a1, load_b(c1, 32, hi), accP);
    accQ = wmma_bf16(a0, load_b(c2, 0, hi), accQ);
    accQ = wmma_bf16(a1, load_b(c2, 32, hi), accQ);
    #pragma unroll
    for (int r = 0; r < 8; ++r) {
      int row = n0 + r + hi * 8;
      P[(size_t)row * CH + nt * 16 + (lane & 15)] = accP[r];
      Q[(size_t)row * CH + nt * 16 + (lane & 15)] = accQ[r];
    }
  }
}

// ---------------------------------------------------------------- Kernel 4: fused KNN (WMMA dist + top-20)
// Block = 128 threads (4 waves). Each wave owns 32 queries (two 16x16 M-tiles),
// so each lane owns exactly one query for the selection phase.
// Candidate tiles (16 pts x 128B = 2KB) are staged in LDS with double-buffered
// global_load_async_to_lds_b128 (one async op per wave per tile, ASYNCcnt-tracked).
// Ranking key = ||c||^2 - 2*q.c   (||q||^2 constant per row, dropped).
__global__ void k_knn(const __bf16* __restrict__ xb,
                      const float* __restrict__ sq,
                      int* __restrict__ nbr) {
  __shared__ __align__(16) float keys[4][32 * 20];   // [wave][row*20+col], 16B-aligned rows
  __shared__ float tk[4][KNN][32];                   // per-lane top-K keys
  __shared__ int   ti[4][KNN][32];                   // per-lane top-K global indices
  __shared__ __align__(16) __bf16 candT[2][16][CH];  // double-buffered candidate tile

  int wave = threadIdx.x >> 5;
  int lane = threadIdx.x & 31;
  int m = lane & 15, hi = lane >> 4;

  int blocksPerBatch = NPTS / 128;                 // 32
  int batch = blockIdx.x / blocksPerBatch;
  int blkIn = blockIdx.x % blocksPerBatch;
  int batchBase = batch * NPTS;
  int qLocalBase = blkIn * 128 + wave * 32;        // first query of this wave
  int qLocal = qLocalBase + lane;                  // lane's own query (local)

  const __bf16* xbatch = xb + (size_t)batchBase * CH;

  // A fragments for the wave's two query tiles, both K chunks, held in VGPRs.
  v16bf aT[2][2];
  #pragma unroll
  for (int t = 0; t < 2; ++t) {
    const __bf16* qrow = xbatch + (size_t)(qLocalBase + t * 16 + m) * CH;
    aT[t][0] = load_a(qrow, 0, hi);
    aT[t][1] = load_a(qrow, 32, hi);
  }

  // init top-K
  #pragma unroll
  for (int j = 0; j < KNN; ++j) { tk[wave][j][lane] = FLT_BIG; ti[wave][j][lane] = 0; }
  float kworst = FLT_BIG;

  // cooperative async prefetch of candidate tile ct: 128 threads x 16B = 2KB
  auto issue = [&](int ct) {
    const __bf16* g = xbatch + (size_t)ct * (16 * CH) + (size_t)threadIdx.x * 8;
    __bf16* l = &candT[ct & 1][0][0] + (size_t)threadIdx.x * 8;
    async_copy_b128(g, l);
  };

  issue(0);
  constexpr int NCT = NPTS / 16;                   // 256 candidate tiles
  for (int ct = 0; ct < NCT; ++ct) {
    int buf = ct & 1;
    bool more = (ct + 1 < NCT);
    if (more) { issue(ct + 1); wait_async_le1(); } else { wait_async_le0(); }
    __syncthreads();                               // candidate tile visible to all waves

    // B fragments from LDS (ds_load_b128 x2 per chunk)
    const __bf16* crow = &candT[buf][lane & 15][0];
    v16bf b0 = load_b(crow, 0, hi);
    v16bf b1 = load_b(crow, 32, hi);
    v8f acc0 = {}; v8f acc1 = {};
    acc0 = wmma_bf16(aT[0][0], b0, acc0);
    acc0 = wmma_bf16(aT[0][1], b1, acc0);
    acc1 = wmma_bf16(aT[1][0], b0, acc1);
    acc1 = wmma_bf16(aT[1][1], b1, acc1);

    int c0 = ct * 16;
    float sqn = sq[batchBase + c0 + (lane & 15)];
    #pragma unroll
    for (int r = 0; r < 8; ++r) {
      int row = r + hi * 8;
      keys[wave][row * 20 + (lane & 15)]        = sqn - 2.0f * acc0[r];
      keys[wave][(16 + row) * 20 + (lane & 15)] = sqn - 2.0f * acc1[r];
    }
    __syncthreads();

    // lane scans the 16 candidates for its own query row (4x float4 DS reads)
    const float4* krow = reinterpret_cast<const float4*>(&keys[wave][lane * 20]);
    #pragma unroll
    for (int tt = 0; tt < 4; ++tt) {
      float4 kv4 = krow[tt];
      float ks[4] = {kv4.x, kv4.y, kv4.z, kv4.w};
      #pragma unroll
      for (int e = 0; e < 4; ++e) {
        int cLoc = c0 + tt * 4 + e;
        float kv = ks[e];
        if (cLoc != qLocal && kv < kworst) {
          // replace current worst entry
          int jm = 0; float km = tk[wave][0][lane];
          #pragma unroll
          for (int j = 1; j < KNN; ++j) {
            float v = tk[wave][j][lane];
            if (v > km) { km = v; jm = j; }
          }
          tk[wave][jm][lane] = kv;
          ti[wave][jm][lane] = batchBase + cLoc;
          float nw = tk[wave][0][lane];
          #pragma unroll
          for (int j = 1; j < KNN; ++j) nw = fmaxf(nw, tk[wave][j][lane]);
          kworst = nw;
        }
      }
    }
    __syncthreads();
  }

  int q = batchBase + qLocal;
  #pragma unroll
  for (int j = 0; j < KNN; ++j) nbr[(size_t)q * KNN + j] = ti[wave][j][lane];
}

// ---------------------------------------------------------------- Kernel 5: BN statistics
// h_e = P[i] + Q[j]; accumulate per-channel sum / sumsq over all E edges.
// 256 threads: o = t&63 (channel), g = t>>6 (center group of 4). 128 centers/block.
__global__ void k_stats(const float* __restrict__ P, const float* __restrict__ Q,
                        const int* __restrict__ nbr,
                        float* __restrict__ statSum, float* __restrict__ statSq) {
  __shared__ float redS[4][CH];
  __shared__ float redQ[4][CH];
  int o = threadIdx.x & 63;
  int g = threadIdx.x >> 6;
  float s = 0.f, s2 = 0.f;
  int base = blockIdx.x * 128;
  for (int it = 0; it < 32; ++it) {
    int j = base + it * 4 + g;
    float qv = Q[(size_t)j * CH + o];
    const int* nj = nbr + (size_t)j * KNN;
    #pragma unroll
    for (int k = 0; k < KNN; ++k) {
      int i = nj[k];
      float h = P[(size_t)i * CH + o] + qv;
      s += h; s2 += h * h;
    }
  }
  redS[g][o] = s; redQ[g][o] = s2;
  __syncthreads();
  if (g == 0) {
    float ts = redS[0][o] + redS[1][o] + redS[2][o] + redS[3][o];
    float tq = redQ[0][o] + redQ[1][o] + redQ[2][o] + redQ[3][o];
    unsafeAtomicAdd(statSum + o, ts);
    unsafeAtomicAdd(statSq + o, tq);
  }
}

// ---------------------------------------------------------------- Kernel 6: BN finalize
__global__ void k_bn_final(const float* __restrict__ statSum, const float* __restrict__ statSq,
                           const float* __restrict__ gamma, const float* __restrict__ beta,
                           float* __restrict__ scale, float* __restrict__ shift) {
  int o = threadIdx.x;
  if (o >= CH) return;
  float invE = 1.0f / (float)((long long)NTOT * KNN);
  float mu = statSum[o] * invE;
  float var = statSq[o] * invE - mu * mu;
  float sc = gamma[o] * rsqrtf(var + BN_EPS);
  scale[o] = sc;
  shift[o] = beta[o] - mu * sc;
}

// ---------------------------------------------------------------- Kernel 7: normalize + ReLU + scatter
__global__ void k_scatter(const float* __restrict__ P, const float* __restrict__ Q,
                          const int* __restrict__ nbr,
                          const float* __restrict__ scale, const float* __restrict__ shift,
                          float* __restrict__ out) {
  int o = threadIdx.x & 63;
  int g = threadIdx.x >> 6;
  float sc = scale[o], sh = shift[o];
  int base = blockIdx.x * 128;
  for (int it = 0; it < 32; ++it) {
    int j = base + it * 4 + g;
    float qv = Q[(size_t)j * CH + o];
    const int* nj = nbr + (size_t)j * KNN;
    #pragma unroll
    for (int k = 0; k < KNN; ++k) {
      int i = nj[k];
      float h = P[(size_t)i * CH + o] + qv;
      float r = fmaxf(sc * h + sh, 0.f);
      unsafeAtomicAdd(out + (size_t)i * CH + o, r);
    }
  }
}

// ---------------------------------------------------------------- launch
extern "C" void kernel_launch(void* const* d_in, const int* in_sizes, int n_in,
                              void* d_out, int out_size, void* d_ws, size_t ws_size,
                              hipStream_t stream) {
  const float* x     = (const float*)d_in[0];
  // d_in[1] = batch (int64) — layout is fixed/contiguous, unused.
  const float* W     = (const float*)d_in[2];
  // d_in[3] = b — cancels exactly inside BatchNorm, unused.
  const float* gamma = (const float*)d_in[4];
  const float* beta  = (const float*)d_in[5];
  float* out = (float*)d_out;

  // workspace layout (256B aligned)
  char* ws = (char*)d_ws;
  constexpr size_t SZ_XB  = (size_t)NTOT * CH * 2;        // 4 MB  bf16 x
  constexpr size_t SZ_SQ  = (size_t)NTOT * 4;             // 128 KB
  constexpr size_t SZ_WB  = (size_t)CH * CH * 2;          // 8 KB each
  constexpr size_t SZ_PQ  = (size_t)NTOT * CH * 4;        // 8 MB each
  constexpr size_t SZ_NBR = (size_t)NTOT * KNN * 4;       // 2.5 MB
  size_t off = 0;
  auto take = [&](size_t bytes) { size_t o = off; off = (off + bytes + 255) & ~(size_t)255; return o; };
  __bf16* xb   = (__bf16*)(ws + take(SZ_XB));
  float*  sq   = (float*) (ws + take(SZ_SQ));
  __bf16* W1b  = (__bf16*)(ws + take(SZ_WB));
  __bf16* W2b  = (__bf16*)(ws + take(SZ_WB));
  float*  P    = (float*) (ws + take(SZ_PQ));
  float*  Q    = (float*) (ws + take(SZ_PQ));
  int*    nbr  = (int*)   (ws + take(SZ_NBR));
  float*  stat = (float*) (ws + take(4 * CH * 4));        // sum | sq | scale | shift
  float* statSum = stat, *statSq = stat + CH, *scale = stat + 2 * CH, *shift = stat + 3 * CH;

  hipMemsetAsync(out, 0, (size_t)out_size * sizeof(float), stream);
  hipMemsetAsync(statSum, 0, 2 * CH * sizeof(float), stream);

  k_prep_points <<<NTOT / 256, 256, 0, stream>>>(x, xb, sq);
  k_prep_weights<<<1, 256, 0, stream>>>(W, W1b, W2b);
  k_feat_gemm   <<<NTOT / 16, 32, 0, stream>>>(xb, W1b, W2b, P, Q);
  k_knn         <<<BATCH * (NPTS / 128), 128, 0, stream>>>(xb, sq, nbr);
  k_stats       <<<NTOT / 128, 256, 0, stream>>>(P, Q, nbr, statSum, statSq);
  k_bn_final    <<<1, 64, 0, stream>>>(statSum, statSq, gamma, beta, scale, shift);
  k_scatter     <<<NTOT / 128, 256, 0, stream>>>(P, Q, nbr, scale, shift, out);
}
